// GCN_79061757985371
// MI455X (gfx1250) — matrix operand order repaired and checked by reading, not compiled, not run on hardware
//
#include <hip/hip_runtime.h>
#include <hip/hip_bf16.h>

#define N_NODES 100000
#define N_EDGES 3200000
#define C_CL    40
#define M_LAB   50000
#define IN_CH   128
#define H_CH    256
#define FC_K    512
#define FC_N    1600
#define BN_EPS  1e-5f

typedef __bf16 bf16_t;
typedef __attribute__((ext_vector_type(16))) bf16_t v16bf;
typedef __attribute__((ext_vector_type(8)))  float  v8f;

// ---------------------------------------------------------------------------
// Degree / normalization
// ---------------------------------------------------------------------------
__global__ __launch_bounds__(256) void deg_kernel(const int* __restrict__ dst,
                                                  float* __restrict__ deg) {
    unsigned e = blockIdx.x * 256u + threadIdx.x;
    if (e < N_EDGES) atomicAdd(&deg[dst[e]], 1.0f);
}

__global__ __launch_bounds__(256) void dinv_kernel(float* __restrict__ deg) {
    unsigned i = blockIdx.x * 256u + threadIdx.x;
    if (i < N_NODES) deg[i] = rsqrtf(deg[i] + 1.0f);   // in-place: deg -> dinv
}

// ---------------------------------------------------------------------------
// Pack B[K,N] (f32 row-major) into bf16 WMMA B-fragment layout:
//   Bp[ (kt*nT16 + nt)*512 + lane*16 + e ]  with
//   k = kt*32 + (e>>3)*16 + (lane>>4)*8 + (e&7),  n = nt*16 + (lane&15)
// One fragment per (ktile,ntile) = 512 bf16 = 1 KB; per-lane 32B contiguous.
// ---------------------------------------------------------------------------
__global__ __launch_bounds__(256) void pack_b_kernel(
    const float* __restrict__ B, bf16_t* __restrict__ Bp, int K, int N)
{
    unsigned o = blockIdx.x * 256u + threadIdx.x;
    if (o >= (unsigned)K * N) return;
    unsigned e    = o & 15u;
    unsigned ln   = (o >> 4) & 31u;
    unsigned tile = o >> 9;
    unsigned nT16 = (unsigned)N >> 4;
    unsigned nt = tile % nT16;
    unsigned kt = tile / nT16;
    unsigned k = kt * 32 + ((e >> 3) << 4) + ((ln >> 4) << 3) + (e & 7u);
    unsigned n = nt * 16 + (ln & 15u);
    Bp[o] = (bf16_t)B[(size_t)k * N + n];
}

// ---------------------------------------------------------------------------
// A fragment: f32 row-major -> bf16 regs per ISA 16-bit A 16x32 layout
// ---------------------------------------------------------------------------
__device__ __forceinline__ v16bf pack_a_bf16(const float* __restrict__ ap) {
    float4 x0 = *(const float4*)(ap);
    float4 x1 = *(const float4*)(ap + 4);
    float4 x2 = *(const float4*)(ap + 16);
    float4 x3 = *(const float4*)(ap + 20);
    v16bf av;
    av[0]=(bf16_t)x0.x;  av[1]=(bf16_t)x0.y;  av[2]=(bf16_t)x0.z;  av[3]=(bf16_t)x0.w;
    av[4]=(bf16_t)x1.x;  av[5]=(bf16_t)x1.y;  av[6]=(bf16_t)x1.z;  av[7]=(bf16_t)x1.w;
    av[8]=(bf16_t)x2.x;  av[9]=(bf16_t)x2.y;  av[10]=(bf16_t)x2.z; av[11]=(bf16_t)x2.w;
    av[12]=(bf16_t)x3.x; av[13]=(bf16_t)x3.y; av[14]=(bf16_t)x3.z; av[15]=(bf16_t)x3.w;
    return av;
}

// ---------------------------------------------------------------------------
// WMMA GEMM: C[M,N] = A[M,K](f32) * Bp(packed bf16) ; one wave = 16x64 tile
// ---------------------------------------------------------------------------
__global__ __launch_bounds__(256) void gemm_kernel(
    const float* __restrict__ A, const bf16_t* __restrict__ Bp,
    float* __restrict__ Cm, int Mrows, int Ncols, int K)
{
    const int lane = threadIdx.x & 31;
    const int wid  = blockIdx.x * 8 + (threadIdx.x >> 5);
    const int nT4  = Ncols >> 6;
    const int nTM  = Mrows >> 4;
    if (wid >= nTM * nT4) return;
    const int nT16 = Ncols >> 4;
    const int rt   = wid / nT4;
    const int ct4  = wid % nT4;
    const int row  = rt * 16 + (lane & 15);
    const int kb   = (lane >> 4) * 8;

    v8f acc[4] = {};
    const float* arow = A + (size_t)row * K;
    for (int k0 = 0; k0 < K; k0 += 32) {
        const float* ap = arow + k0 + kb;
        if (k0 + 32 < K) __builtin_prefetch(ap + 32, 0, 1);
        v16bf av = pack_a_bf16(ap);
        const bf16_t* bpb = Bp + (((size_t)(k0 >> 5) * nT16 + ct4 * 4) * 32 + lane) * 16;
#pragma unroll
        for (int t = 0; t < 4; ++t) {
            v16bf bv = *(const v16bf*)(bpb + (size_t)t * 512);
            acc[t] = __builtin_amdgcn_wmma_f32_16x16x32_bf16(
                false, av, false, bv, (short)0, acc[t], false, false);
        }
    }
    const int srow = rt * 16 + ((lane >> 4) ? 8 : 0);
#pragma unroll
    for (int t = 0; t < 4; ++t) {
        int c = ct4 * 64 + t * 16 + (lane & 15);
#pragma unroll
        for (int v = 0; v < 8; ++v)
            Cm[(size_t)(srow + v) * Ncols + c] = acc[t][v];
    }
}

// ---------------------------------------------------------------------------
// agg init: agg = dinv^2 * t + bias  (self loop + conv bias)
// ---------------------------------------------------------------------------
__global__ __launch_bounds__(256) void init_agg_kernel(
    const float* __restrict__ t, const float* __restrict__ dinv,
    const float* __restrict__ bias, float* __restrict__ agg)
{
    unsigned i = blockIdx.x * 256u + threadIdx.x;       // N*H = 25.6M
    if (i >= (unsigned)N_NODES * H_CH) return;
    unsigned r = i >> 8, c = i & 255u;
    float dv = dinv[r];
    agg[i] = dv * dv * t[i] + bias[c];
}

// ---------------------------------------------------------------------------
// Edge scatter: agg[dst] += t[src] * dinv[src]*dinv[dst]   (atomic f32)
// ---------------------------------------------------------------------------
__global__ __launch_bounds__(256) void edge_agg_kernel(
    const int* __restrict__ src, const int* __restrict__ dst,
    const float* __restrict__ dinv, const float* __restrict__ t,
    float* __restrict__ agg)
{
    unsigned idx = blockIdx.x * 256u + threadIdx.x;     // E*64 = 204.8M
    unsigned e = idx >> 6;
    if (e >= N_EDGES) return;
    unsigned q = (idx & 63u) << 2;
    int s = src[e], d = dst[e];
    float w = dinv[s] * dinv[d];
    float4 v = *(const float4*)(t + (size_t)s * H_CH + q);
    float* o = agg + (size_t)d * H_CH + q;
    atomicAdd(o + 0, v.x * w);
    atomicAdd(o + 1, v.y * w);
    atomicAdd(o + 2, v.z * w);
    atomicAdd(o + 3, v.w * w);
}

// ---------------------------------------------------------------------------
// BatchNorm: column sums / sum-of-squares, then normalize + ReLU in place
// ---------------------------------------------------------------------------
__global__ __launch_bounds__(256) void bn_stats_kernel(
    const float* __restrict__ h, float* __restrict__ sums)
{
    int col = threadIdx.x;                               // H_CH = 256 = blockDim
    int r0 = blockIdx.x * 128;
    int r1 = r0 + 128; if (r1 > N_NODES) r1 = N_NODES;
    float s = 0.f, ss = 0.f;
    for (int r = r0; r < r1; ++r) {
        float v = h[(size_t)r * H_CH + col];
        s += v; ss += v * v;
    }
    atomicAdd(&sums[col], s);
    atomicAdd(&sums[H_CH + col], ss);
}

__global__ __launch_bounds__(256) void bn_apply_kernel(
    float* __restrict__ h, const float* __restrict__ sums,
    const float* __restrict__ g, const float* __restrict__ be)
{
    unsigned i = blockIdx.x * 256u + threadIdx.x;
    if (i >= (unsigned)N_NODES * H_CH) return;
    unsigned c = i & 255u;
    const float inv_n = 1.0f / (float)N_NODES;
    float mean = sums[c] * inv_n;
    float var  = sums[H_CH + c] * inv_n - mean * mean;
    float y = g[c] * (h[i] - mean) * rsqrtf(var + BN_EPS) + be[c];
    h[i] = fmaxf(y, 0.0f);
}

// ---------------------------------------------------------------------------
// Cluster pooling
// ---------------------------------------------------------------------------
__global__ __launch_bounds__(256) void assign_kernel(
    const float* __restrict__ cluster_id, int* __restrict__ assign,
    float* __restrict__ counts)
{
    unsigned m = blockIdx.x * 256u + threadIdx.x;
    if (m >= M_LAB) return;
    const float* row = cluster_id + (size_t)m * C_CL;
    int best = 0; float bv = row[0];
    for (int c = 1; c < C_CL; ++c) {
        float v = row[c];
        if (v > bv) { bv = v; best = c; }
    }
    assign[m] = best;
    atomicAdd(&counts[best], 1.0f);
}

__global__ __launch_bounds__(256) void cfsum_kernel(
    const float* __restrict__ h, const int* __restrict__ ci,
    const int* __restrict__ assign, float* __restrict__ cfsum)
{
    unsigned idx = blockIdx.x * 256u + threadIdx.x;     // M*32 = 1.6M
    unsigned m = idx >> 5;
    if (m >= M_LAB) return;
    unsigned c8 = (idx & 31u) * 8;
    int a = assign[m];
    const float* src = h + (size_t)ci[m] * H_CH + c8;
    float* dstp = cfsum + (size_t)a * H_CH + c8;
#pragma unroll
    for (int j = 0; j < 8; ++j) atomicAdd(dstp + j, src[j]);
}

__global__ __launch_bounds__(256) void cf_div_kernel(
    const float* __restrict__ cfsum, const float* __restrict__ counts,
    float* __restrict__ cf)
{
    unsigned i = blockIdx.x * 256u + threadIdx.x;       // C*H = 10240
    if (i >= (unsigned)C_CL * H_CH) return;
    float cnt = counts[i >> 8];
    cf[i] = cfsum[i] / fmaxf(cnt, 1.0f);
}

// ---------------------------------------------------------------------------
// FC: out[2M,1600] = concat-rows @ fcW + fcb; A rows resolved via indices,
// fcW consumed pre-packed (bf16 WMMA fragments).
// ---------------------------------------------------------------------------
__global__ __launch_bounds__(256) void fc_gemm_kernel(
    const float* __restrict__ h, const float* __restrict__ cf,
    const int* __restrict__ ci, const int* __restrict__ assign,
    const bf16_t* __restrict__ Wp, const float* __restrict__ fcb,
    float* __restrict__ out)
{
    const int lane = threadIdx.x & 31;
    const int wid  = blockIdx.x * 8 + (threadIdx.x >> 5);
    const int nT4  = FC_N / 64;                   // 25
    const int nTM  = (2 * M_LAB) / 16;            // 6250
    if (wid >= nTM * nT4) return;
    const int nT16 = FC_N / 16;                   // 100
    const int rt   = wid / nT4;
    const int ct4  = wid % nT4;
    const int row  = rt * 16 + (lane & 15);
    const int kb   = (lane >> 4) * 8;

    const float *p0, *p1;
    if (row < M_LAB) {
        p0 = h  + (size_t)ci[row]     * H_CH;
        p1 = cf + (size_t)assign[row] * H_CH;
    } else {
        int m = row - M_LAB;
        p0 = cf + (size_t)assign[m] * H_CH;
        p1 = h  + (size_t)ci[m]     * H_CH;
    }

    v8f acc[4] = {};
    for (int k0 = 0; k0 < FC_K; k0 += 32) {
        const float* ap = ((k0 < H_CH) ? p0 : p1) + (k0 & (H_CH - 1)) + kb;
        v16bf av = pack_a_bf16(ap);
        const bf16_t* bpb = Wp + (((size_t)(k0 >> 5) * nT16 + ct4 * 4) * 32 + lane) * 16;
#pragma unroll
        for (int t = 0; t < 4; ++t) {
            v16bf bv = *(const v16bf*)(bpb + (size_t)t * 512);
            acc[t] = __builtin_amdgcn_wmma_f32_16x16x32_bf16(
                false, av, false, bv, (short)0, acc[t], false, false);
        }
    }
    const int srow = rt * 16 + ((lane >> 4) ? 8 : 0);
#pragma unroll
    for (int t = 0; t < 4; ++t) {
        int c = ct4 * 64 + t * 16 + (lane & 15);
        float bc = fcb[c];
#pragma unroll
        for (int v = 0; v < 8; ++v)
            out[(size_t)(srow + v) * FC_N + c] = acc[t][v] + bc;
    }
}

// ---------------------------------------------------------------------------
// Launch
// ---------------------------------------------------------------------------
extern "C" void kernel_launch(void* const* d_in, const int* in_sizes, int n_in,
                              void* d_out, int out_size, void* d_ws, size_t ws_size,
                              hipStream_t stream)
{
    const float* x          = (const float*)d_in[0];
    const int*   ei         = (const int*)  d_in[1];     // [2, E]
    const float* cluster_id = (const float*)d_in[2];     // [M, C]
    const int*   ci         = (const int*)  d_in[3];     // [M]
    const float* W[3]  = { (const float*)d_in[4],  (const float*)d_in[8],  (const float*)d_in[12] };
    const float* b[3]  = { (const float*)d_in[5],  (const float*)d_in[9],  (const float*)d_in[13] };
    const float* g[3]  = { (const float*)d_in[6],  (const float*)d_in[10], (const float*)d_in[14] };
    const float* be[3] = { (const float*)d_in[7],  (const float*)d_in[11], (const float*)d_in[15] };
    const float* fcW = (const float*)d_in[16];
    const float* fcb = (const float*)d_in[17];
    float* out = (float*)d_out;

    const int* e_src = ei;
    const int* e_dst = ei + N_EDGES;

    // workspace carve-up (256B aligned)
    char* ws = (char*)d_ws;
    size_t off = 0;
    auto take = [&](size_t bytes) { void* p = ws + off; off = (off + bytes + 255) & ~(size_t)255; return p; };
    float*  tbuf   = (float*)take((size_t)N_NODES * H_CH * 4);   // h @ W
    float*  agg    = (float*)take((size_t)N_NODES * H_CH * 4);   // aggregated / h
    float*  dinv   = (float*)take((size_t)N_NODES * 4);
    float*  bnsums = (float*)take(2 * H_CH * 4);
    int*    assign = (int*)  take((size_t)M_LAB * 4);
    float*  counts = (float*)take(C_CL * 4);
    float*  cfsum  = (float*)take((size_t)C_CL * H_CH * 4);
    float*  cf     = (float*)take((size_t)C_CL * H_CH * 4);
    bf16_t* Wp[3];
    Wp[0] = (bf16_t*)take((size_t)IN_CH * H_CH * 2);
    Wp[1] = (bf16_t*)take((size_t)H_CH * H_CH * 2);
    Wp[2] = (bf16_t*)take((size_t)H_CH * H_CH * 2);
    bf16_t* fcWp = (bf16_t*)take((size_t)FC_K * FC_N * 2);
    (void)ws_size; (void)n_in; (void)in_sizes; (void)out_size;

    // pre-pack all weight matrices into WMMA bf16 fragments
    pack_b_kernel<<<(IN_CH * H_CH + 255) / 256, 256, 0, stream>>>(W[0], Wp[0], IN_CH, H_CH);
    pack_b_kernel<<<(H_CH * H_CH + 255) / 256, 256, 0, stream>>>(W[1], Wp[1], H_CH, H_CH);
    pack_b_kernel<<<(H_CH * H_CH + 255) / 256, 256, 0, stream>>>(W[2], Wp[2], H_CH, H_CH);
    pack_b_kernel<<<(FC_K * FC_N + 255) / 256, 256, 0, stream>>>(fcW, fcWp, FC_K, FC_N);

    // degrees -> dinv
    hipMemsetAsync(dinv, 0, (size_t)N_NODES * 4, stream);
    deg_kernel<<<(N_EDGES + 255) / 256, 256, 0, stream>>>(e_dst, dinv);
    dinv_kernel<<<(N_NODES + 255) / 256, 256, 0, stream>>>(dinv);

    const unsigned nh_blocks   = ((unsigned)N_NODES * H_CH + 255) / 256;  // 100000
    const unsigned edge_blocks = ((unsigned)N_EDGES * 64 + 255) / 256;    // 800000
    const unsigned bn_blocks   = (N_NODES + 127) / 128;                   // 782

    for (int l = 0; l < 3; ++l) {
        const float* A = (l == 0) ? x : agg;
        const int K = (l == 0) ? IN_CH : H_CH;
        const int waves = (N_NODES / 16) * (H_CH / 64);                   // 25000
        gemm_kernel<<<(waves + 7) / 8, 256, 0, stream>>>(
            A, Wp[l], tbuf, N_NODES, H_CH, K);
        init_agg_kernel<<<nh_blocks, 256, 0, stream>>>(tbuf, dinv, b[l], agg);
        edge_agg_kernel<<<edge_blocks, 256, 0, stream>>>(e_src, e_dst, dinv, tbuf, agg);
        hipMemsetAsync(bnsums, 0, 2 * H_CH * 4, stream);
        bn_stats_kernel<<<bn_blocks, 256, 0, stream>>>(agg, bnsums);
        bn_apply_kernel<<<nh_blocks, 256, 0, stream>>>(agg, bnsums, g[l], be[l]);
    }

    // cluster pooling
    hipMemsetAsync(counts, 0, C_CL * 4, stream);
    hipMemsetAsync(cfsum, 0, (size_t)C_CL * H_CH * 4, stream);
    assign_kernel<<<(M_LAB + 255) / 256, 256, 0, stream>>>(cluster_id, assign, counts);
    cfsum_kernel<<<((unsigned)M_LAB * 32 + 255) / 256, 256, 0, stream>>>(agg, ci, assign, cfsum);
    cf_div_kernel<<<(C_CL * H_CH + 255) / 256, 256, 0, stream>>>(cfsum, counts, cf);

    // final FC
    const int fc_waves = ((2 * M_LAB) / 16) * (FC_N / 64);                // 156250
    fc_gemm_kernel<<<(fc_waves + 7) / 8, 256, 0, stream>>>(
        agg, cf, ci, assign, fcWp, fcb, out);
}
